// MambaDualScan_36060545417601
// MI455X (gfx1250) — matrix operand order repaired and checked by reading, not compiled, not run on hardware
//
#include <hip/hip_runtime.h>
#include <hip/hip_bf16.h>

// ---------------- problem constants ----------------
constexpr int kB    = 16;
constexpr int kT    = 24;
constexpr int kNseq = 207;
constexpr int kC    = 256;               // c
constexpr int kS    = kB * kNseq;        // 3312 sequences
constexpr int kM    = kS * kT;           // 79488 token rows (multiple of 16)
constexpr int kIn   = 2 * kC;            // 512
constexpr int kRed  = kC / 2;            // 128
constexpr int kDI   = 2 * kRed;          // 256 d_inner
constexpr int kDS   = 16;                // D_STATE
constexpr int kDR   = 8;                 // dt_rank
constexpr int kXP   = 48;                // padded x_dbl width (40 -> 48)
constexpr int kKDT  = 32;                // padded dt-GEMM K (8 -> 32)

typedef __bf16 bf16;
typedef __attribute__((ext_vector_type(8)))  __bf16 v8bf;
typedef __attribute__((ext_vector_type(16))) __bf16 v16bf;
typedef __attribute__((ext_vector_type(8)))  float  v8f;

// ---------------- WMMA fragment loads (pure bf16, unconditional, vectorized) ----------------
// A fragment 16x32: lanes 0-15 -> M=lane, K in {0..7,16..23}; lanes 16-31 -> M=lane-16, K in {8..15,24..31}
static __device__ __forceinline__ v16bf load_a_frag(const bf16* __restrict__ A, int lda,
                                                    int m0, int k0, int lane) {
  int m    = m0 + (lane & 15);
  int half = lane >> 4;
  const bf16* p = A + (long)m * lda + k0 + half * 8;
  v8bf lo = *(const v8bf*)(p);        // K = k0+half*8   .. +7
  v8bf hi = *(const v8bf*)(p + 16);   // K = k0+16+half*8.. +7
  return __builtin_shufflevector(lo, hi, 0,1,2,3,4,5,6,7,8,9,10,11,12,13,14,15);
}

// B fragment 32x16 from W[N][K] row-major: lane -> N=lane&15, K in {k0+half*8..+7, k0+16+half*8..+7}
static __device__ __forceinline__ v16bf load_b_frag(const bf16* __restrict__ W, int ldw,
                                                    int n0, int k0, int lane) {
  int n    = n0 + (lane & 15);
  int half = lane >> 4;
  const bf16* p = W + (long)n * ldw + k0 + half * 8;
  v8bf lo = *(const v8bf*)(p);
  v8bf hi = *(const v8bf*)(p + 16);
  return __builtin_shufflevector(lo, hi, 0,1,2,3,4,5,6,7,8,9,10,11,12,13,14,15);
}

// ---------------- generic WMMA GEMM: C[MxN] = A[MxK] @ W[NxK]^T (+bias)(+act) ----------------
// Each wave computes a 16 x (NB*16) output strip: A fragment reused NB times per K-step.
// mode 0: fp32 store | 1: fp32 accumulate | 2: fp32 scatter to (b,t,n,c) | 3: bf16 store | 4: fp32+bf16 dual
// act  0: none | 1: softplus
template <int K, int NB>
__global__ void gemm_bf16(const bf16* __restrict__ A, int lda,
                          const bf16* __restrict__ W, int ldw,
                          float* __restrict__ Cf, bf16* __restrict__ Cbf,
                          int ldc, const float* __restrict__ bias,
                          int M, int N, int mode, int act) {
  int lane = threadIdx.x & 31;
  int wave = threadIdx.x >> 5;
  int tilesM = M >> 4;                       // M multiple of 16
  int ngrp   = (N >> 4) / NB;                // (N/16) guaranteed multiple of NB
  int tile = blockIdx.x * 8 + wave;
  if (tile >= tilesM * ngrp) return;         // wave-uniform exit: EXEC all-ones for WMMA
  int m0 = (tile % tilesM) << 4;
  int n0 = (tile / tilesM) * (NB * 16);

  v8f acc[NB];
#pragma unroll
  for (int j = 0; j < NB; ++j) acc[j] = (v8f)0.f;

#pragma unroll
  for (int k0 = 0; k0 < K; k0 += 32) {
    v16bf a = load_a_frag(A, lda, m0, k0, lane);
#pragma unroll
    for (int j = 0; j < NB; ++j) {
      v16bf b = load_b_frag(W, ldw, n0 + j * 16, k0, lane);
      acc[j] = __builtin_amdgcn_wmma_f32_16x16x32_bf16(false, a, false, b, (short)0,
                                                       acc[j], false, false);
    }
  }

  int half  = lane >> 4;
  int mbase = m0 + half * 8;
#pragma unroll
  for (int j = 0; j < NB; ++j) {
    int n = n0 + j * 16 + (lane & 15);
    float bv = (bias != nullptr) ? bias[n] : 0.f;
    float v[8];
#pragma unroll
    for (int i = 0; i < 8; ++i) {
      float t = acc[j][i] + bv;
      if (act == 1) t = (t > 20.f) ? t : log1pf(__expf(t));   // softplus
      v[i] = t;
    }
    if (mode == 0) {
#pragma unroll
      for (int i = 0; i < 8; ++i) Cf[(long)(mbase + i) * ldc + n] = v[i];
    } else if (mode == 1) {
#pragma unroll
      for (int i = 0; i < 8; ++i) Cf[(long)(mbase + i) * ldc + n] += v[i];
    } else if (mode == 2) {
#pragma unroll
      for (int i = 0; i < 8; ++i) {
        int m = mbase + i;
        int s = m / kT, t = m % kT;
        int bb = s / kNseq, nn = s % kNseq;
        Cf[(((long)bb * kT + t) * kNseq + nn) * kC + n] = v[i];
      }
    } else if (mode == 3) {
#pragma unroll
      for (int i = 0; i < 8; ++i) Cbf[(long)(mbase + i) * ldc + n] = (bf16)v[i];
    } else {
#pragma unroll
      for (int i = 0; i < 8; ++i) {
        Cf[(long)(mbase + i) * ldc + n]  = v[i];
        Cbf[(long)(mbase + i) * ldc + n] = (bf16)v[i];
      }
    }
  }
}

// ---------------- weight pad + fp32->bf16 convert ----------------
__global__ void cvt_pad_bf16(const float* __restrict__ src, bf16* __restrict__ dst,
                             int Ns, int Ks, int Nd, int Kd) {
  int idx = blockIdx.x * 256 + threadIdx.x;
  if (idx >= Nd * Kd) return;
  int n = idx / Kd, k = idx % Kd;
  float v = (n < Ns && k < Ks) ? src[n * Ks + k] : 0.f;
  dst[idx] = (bf16)v;
}

// ---------------- LN1 over concat(x,qk) -> hcat[M x 512] (bf16) ----------------
__global__ void ln1_concat(const float* __restrict__ x, const float* __restrict__ qk,
                           const float* __restrict__ g, const float* __restrict__ b,
                           bf16* __restrict__ hcat) {
  int lane = threadIdx.x & 31;
  int r = blockIdx.x * 8 + (threadIdx.x >> 5);
  int s = r / kT, t = r % kT;
  int bb = s / kNseq, nn = s % kNseq;
  long base = (((long)bb * kT + t) * kNseq + nn) * kC;

  float vals[16];
  float sum = 0.f, sq = 0.f;
#pragma unroll
  for (int i = 0; i < 16; ++i) {
    int e = i * 32 + lane;
    float v = (e < kC) ? x[base + e] : qk[base + e - kC];
    vals[i] = v;
    sum += v; sq += v * v;
  }
#pragma unroll
  for (int off = 16; off >= 1; off >>= 1) {
    sum += __shfl_xor(sum, off, 32);
    sq  += __shfl_xor(sq,  off, 32);
  }
  float mu = sum * (1.f / kIn);
  float var = sq * (1.f / kIn) - mu * mu;
  float rstd = rsqrtf(var + 1e-5f);
#pragma unroll
  for (int i = 0; i < 16; ++i) {
    int e = i * 32 + lane;
    hcat[(long)r * kIn + e] = (bf16)((vals[i] - mu) * rstd * g[e] + b[e]);
  }
}

// ---------------- LN2 over hsum[M x 128] (fp32) -> hn (bf16) ----------------
__global__ void ln2_rows(const float* __restrict__ hin, const float* __restrict__ g,
                         const float* __restrict__ b, bf16* __restrict__ hout) {
  int lane = threadIdx.x & 31;
  int r = blockIdx.x * 8 + (threadIdx.x >> 5);
  const float* row = hin + (long)r * kRed;
  float vals[4];
  float sum = 0.f, sq = 0.f;
#pragma unroll
  for (int i = 0; i < 4; ++i) {
    int e = i * 32 + lane;
    float v = row[e];
    vals[i] = v;
    sum += v; sq += v * v;
  }
#pragma unroll
  for (int off = 16; off >= 1; off >>= 1) {
    sum += __shfl_xor(sum, off, 32);
    sq  += __shfl_xor(sq,  off, 32);
  }
  float mu = sum * (1.f / kRed);
  float var = sq * (1.f / kRed) - mu * mu;
  float rstd = rsqrtf(var + 1e-5f);
#pragma unroll
  for (int i = 0; i < 4; ++i) {
    int e = i * 32 + lane;
    hout[(long)r * kRed + e] = (bf16)((vals[i] - mu) * rstd * g[e] + b[e]);
  }
}

// ---------------- causal conv (K=4) + SiLU over x-half of xz (bf16 in, f32+bf16 out) ----------------
// dir=0: taps t-3..t (forward);  dir=1: taps t..t+3 (time-reversed sequence)
__global__ void conv_silu(const bf16* __restrict__ xz, const float* __restrict__ cw,
                          const float* __restrict__ cb, float* __restrict__ uf,
                          bf16* __restrict__ ubf, int dir) {
  long idx = (long)blockIdx.x * 256 + threadIdx.x;   // grid = kM blocks
  int ch = (int)(idx & 255);
  int r  = (int)(idx >> 8);
  int s = r / kT, t = r % kT;
  float acc = cb[ch];
#pragma unroll
  for (int k = 0; k < 4; ++k) {
    int tt = dir ? (t + 3 - k) : (t - 3 + k);
    if (tt >= 0 && tt < kT)
      acc += (float)xz[((long)s * kT + tt) * kIn + ch] * cw[ch * 4 + k];
  }
  float sv = acc / (1.f + __expf(-acc));   // silu
  uf[idx]  = sv;
  ubf[idx] = (bf16)sv;
}

// ---------------- selective scan: block per sequence, thread per channel ----------------
__global__ void selective_scan(const float* __restrict__ u, const float* __restrict__ dt,
                               const float* __restrict__ xdbl, const float* __restrict__ Alog,
                               const float* __restrict__ Dp, float* __restrict__ ys, int dir) {
  __shared__ float bc[2 * kDS];      // B[16] then C[16] for the current step
  int s  = blockIdx.x;
  int ch = threadIdx.x;

  float A[kDS];
#pragma unroll
  for (int j = 0; j < kDS; ++j) A[j] = -__expf(Alog[ch * kDS + j]);
  float Dch = Dp[ch];
  float h[kDS];
#pragma unroll
  for (int j = 0; j < kDS; ++j) h[j] = 0.f;

  for (int step = 0; step < kT; ++step) {
    int t = dir ? (kT - 1 - step) : step;
    long r = (long)s * kT + t;
    __syncthreads();
    if (ch < 2 * kDS) bc[ch] = xdbl[r * kXP + kDR + ch];
    __syncthreads();
    float dtv = dt[r * kDI + ch];
    float uv  = u[r * kDI + ch];
    float du  = dtv * uv;
    float y = 0.f;
#pragma unroll
    for (int j = 0; j < kDS; ++j) {
      h[j] = h[j] * __expf(dtv * A[j]) + du * bc[j];
      y += h[j] * bc[kDS + j];
    }
    ys[r * kDI + ch] = y + uv * Dch;
  }
}

// ---------------- gate: gated = ys * silu(z)  (bf16 out for W_out GEMM) ----------------
__global__ void gate_mul(const float* __restrict__ ys, const bf16* __restrict__ xz,
                         bf16* __restrict__ gated) {
  long idx = (long)blockIdx.x * 256 + threadIdx.x;
  int ch = (int)(idx & 255);
  long r = idx >> 8;
  float z = (float)xz[r * kIn + kDI + ch];
  gated[idx] = (bf16)(ys[idx] * (z / (1.f + __expf(-z))));
}

// ---------------- host-side GEMM launcher ----------------
template <int K, int NB>
static void launch_gemm(const bf16* A, int lda, const bf16* W, int ldw,
                        float* Cf, bf16* Cbf, int ldc, const float* bias,
                        int M, int N, int mode, int act, hipStream_t stream) {
  int tiles = (M >> 4) * ((N >> 4) / NB);
  int blocks = (tiles + 7) / 8;
  gemm_bf16<K, NB><<<blocks, 256, 0, stream>>>(A, lda, W, ldw, Cf, Cbf, ldc, bias, M, N, mode, act);
}

static void launch_cvt(const float* src, bf16* dst, int Ns, int Ks, int Nd, int Kd,
                       hipStream_t stream) {
  int n = Nd * Kd;
  cvt_pad_bf16<<<(n + 255) / 256, 256, 0, stream>>>(src, dst, Ns, Ks, Nd, Kd);
}

extern "C" void kernel_launch(void* const* d_in, const int* in_sizes, int n_in,
                              void* d_out, int out_size, void* d_ws, size_t ws_size,
                              hipStream_t stream) {
  (void)in_sizes; (void)n_in; (void)out_size; (void)ws_size;
  const float* x   = (const float*)d_in[0];
  const float* qk  = (const float*)d_in[1];
  const float* g1  = (const float*)d_in[2];
  const float* b1  = (const float*)d_in[3];
  const float* W1  = (const float*)d_in[4];
  const float* bl1 = (const float*)d_in[5];
  const float* g2  = (const float*)d_in[24];
  const float* b2  = (const float*)d_in[25];
  const float* W2  = (const float*)d_in[26];
  const float* bl2 = (const float*)d_in[27];
  float* out = (float*)d_out;

  // workspace carve-out (256B aligned)
  char* ws = (char*)d_ws;
  size_t off = 0;
  auto alloc = [&](size_t bytes) -> void* {
    off = (off + 255) & ~(size_t)255;
    void* p = ws + off;
    off += bytes;
    return p;
  };
  const long M = kM;
  // bf16 weight copies (padded where irregular)
  bf16* wW1   = (bf16*)alloc((size_t)kRed * kIn * 2);
  bf16* wWin[2]  = { (bf16*)alloc((size_t)kIn * kRed * 2),  (bf16*)alloc((size_t)kIn * kRed * 2) };
  bf16* wWx[2]   = { (bf16*)alloc((size_t)kXP * kDI * 2),   (bf16*)alloc((size_t)kXP * kDI * 2) };
  bf16* wWdt[2]  = { (bf16*)alloc((size_t)kDI * kKDT * 2),  (bf16*)alloc((size_t)kDI * kKDT * 2) };
  bf16* wWout[2] = { (bf16*)alloc((size_t)kRed * kDI * 2),  (bf16*)alloc((size_t)kRed * kDI * 2) };
  bf16* wW2   = (bf16*)alloc((size_t)kC * kRed * 2);
  // activations
  bf16*  hcat = (bf16*)alloc((size_t)M * kIn * 2);    // LN'd concat (bf16)
  bf16*  hred = (bf16*)alloc((size_t)M * kRed * 2);   // 128-wide (reused as hn)
  bf16*  xz   = (bf16*)alloc((size_t)M * kIn * 2);    // W_in output (bf16)
  float* uf   = (float*)alloc((size_t)M * kDI * 4);   // conv+silu (fp32, scan input)
  bf16*  ubf  = (bf16*)alloc((size_t)M * kDI * 2);    // conv+silu (bf16), reused as gated y
  float* xdbl = (float*)alloc((size_t)M * kXP * 4);   // fp32, stride 48 (B/C for scan)
  bf16*  xdbf = (bf16*)alloc((size_t)M * kXP * 2);    // bf16 copy (dt GEMM A)
  float* dtb  = (float*)alloc((size_t)M * kDI * 4);   // softplus(dt)
  float* ysb  = (float*)alloc((size_t)M * kDI * 4);   // scan output
  float* hsum = (float*)alloc((size_t)M * kRed * 4);  // fw + bw

  // 0) convert/pad weights to bf16
  launch_cvt(W1, wW1, kRed, kIn, kRed, kIn, stream);
  launch_cvt(W2, wW2, kC, kRed, kC, kRed, stream);
  for (int dir = 0; dir < 2; ++dir) {
    int base = 6 + 9 * dir;
    launch_cvt((const float*)d_in[base + 0], wWin[dir],  kIn, kRed, kIn, kRed, stream);
    launch_cvt((const float*)d_in[base + 3], wWx[dir],   40,  kDI,  kXP, kDI,  stream);
    launch_cvt((const float*)d_in[base + 4], wWdt[dir],  kDI, kDR,  kDI, kKDT, stream);
    launch_cvt((const float*)d_in[base + 8], wWout[dir], kRed, kDI, kRed, kDI, stream);
  }

  // 1) LN over concat(x,qk), sequence-major layout
  ln1_concat<<<kM / 8, 256, 0, stream>>>(x, qk, g1, b1, hcat);

  // 2) hred = hcat @ W1^T + bl1   (K=512, N=128) -> bf16
  launch_gemm<kIn, 4>(hcat, kIn, wW1, kIn, nullptr, hred, kRed, bl1, kM, kRed, 3, 0, stream);

  // 3) both mamba directions
  for (int dir = 0; dir < 2; ++dir) {
    int base = 6 + 9 * dir;
    const float* cw   = (const float*)d_in[base + 1];  // (256,4)
    const float* cb   = (const float*)d_in[base + 2];  // (256)
    const float* bdt  = (const float*)d_in[base + 5];  // (256)
    const float* Alog = (const float*)d_in[base + 6];  // (256,16)
    const float* Dp   = (const float*)d_in[base + 7];  // (256)

    // xz = hred @ W_in^T            (K=128, N=512) -> bf16
    launch_gemm<kRed, 4>(hred, kRed, wWin[dir], kRed, nullptr, xz, kIn, nullptr, kM, kIn, 3, 0, stream);
    // u = silu(causal_conv(x-half)) -> fp32 + bf16
    conv_silu<<<kM, 256, 0, stream>>>(xz, cw, cb, uf, ubf, dir);
    // xdbl = u @ W_x^T              (K=256, N=48 padded, NB=3) -> fp32 + bf16
    launch_gemm<kDI, 3>(ubf, kDI, wWx[dir], kDI, xdbl, xdbf, kXP, nullptr, kM, kXP, 4, 0, stream);
    // dt = softplus(xdbl[:, :8] @ W_dt^T + b_dt)   (K=32 padded-zero, N=256) -> fp32
    launch_gemm<kKDT, 4>(xdbf, kXP, wWdt[dir], kKDT, dtb, nullptr, kDI, bdt, kM, kDI, 0, 1, stream);
    // selective scan
    selective_scan<<<kS, 256, 0, stream>>>(uf, dtb, xdbl, Alog, Dp, ysb, dir);
    // gated (into ubf): ys * silu(z)
    gate_mul<<<kM, 256, 0, stream>>>(ysb, xz, ubf);
    // hsum (+)= gated @ W_out^T     (K=256, N=128) -> fp32
    launch_gemm<kDI, 4>(ubf, kDI, wWout[dir], kDI, hsum, nullptr, kRed, nullptr, kM, kRed,
                        dir == 0 ? 0 : 1, 0, stream);
  }

  // 4) LN2 (-> hred as bf16), then out = hn @ W2^T + bl2 scattered to (b,t,n,c)
  ln2_rows<<<kM / 8, 256, 0, stream>>>(hsum, g2, b2, hred);
  launch_gemm<kRed, 4>(hred, kRed, wW2, kRed, out, nullptr, kC, bl2, kM, kC, 2, 0, stream);
}
